// MySelfAttention_1073741824308
// MI455X (gfx1250) — matrix operand order repaired and checked by reading, not compiled, not run on hardware
//
#include <hip/hip_runtime.h>
#include <hip/hip_bf16.h>

#define BN 4
#define SN 2048
#define DN 768

typedef __bf16 bf16_t;
typedef bf16_t v16bf __attribute__((ext_vector_type(16)));
typedef bf16_t v8bf  __attribute__((ext_vector_type(8)));
typedef float  v8f   __attribute__((ext_vector_type(8)));

// ---------------------------------------------------------------------------
// Fragment loaders matching CDNA5 16-bit WMMA VGPR layouts (ISA 05_wmma §7.12.2)
// A (16x32, MxK): lane l<16 -> row l, K = {k0..k0+7, k0+16..k0+23}
//                 lane l>=16 -> row l-16, K = {k0+8..k0+15, k0+24..k0+31}
__device__ __forceinline__ v16bf load_frag_a(const bf16_t* base, int ld, int lane) {
    const int half = (lane >> 4) & 1;
    const int r    = lane & 15;
    const bf16_t* p = base + (size_t)r * ld + half * 8;
    v8bf lo = *(const v8bf*)(p);
    v8bf hi = *(const v8bf*)(p + 16);
    return __builtin_shufflevector(lo, hi, 0,1,2,3,4,5,6,7,8,9,10,11,12,13,14,15);
}

// B (32x16, KxN): lane n<16 -> col n, K = {k0..k0+7, k0+16..k0+23} ; lane n>=16 -> col n-16, K = {k0+8..k0+15, k0+24..k0+31}
// source: column n contiguous, i.e. src[n][k], stride ld; base at (n0, k0).
__device__ __forceinline__ v16bf load_frag_b(const bf16_t* base, int ld, int lane) {
    const int half = (lane >> 4) & 1;
    const int n    = lane & 15;
    const bf16_t* p = base + (size_t)n * ld + half * 8;
    v8bf lo = *(const v8bf*)(p);
    v8bf hi = *(const v8bf*)(p + 16);
    return __builtin_shufflevector(lo, hi, 0,1,2,3,4,5,6,7,8,9,10,11,12,13,14,15);
}

__device__ __forceinline__ v8f wmma_bf16(v16bf a, v16bf b, v8f c) {
    return __builtin_amdgcn_wmma_f32_16x16x32_bf16(false, a, false, b, (short)0, c, false, false);
}

// ---------------------------------------------------------------------------
// Stage 0a: fp32 -> bf16 elementwise
__global__ void cvt_bf16_kernel(const float* __restrict__ src, bf16_t* __restrict__ dst, int n) {
    for (int i = blockIdx.x * blockDim.x + threadIdx.x; i < n; i += gridDim.x * blockDim.x)
        dst[i] = (bf16_t)src[i];
}

// Stage 0b: Wt[e][d] = (bf16) W[d][e]   (D x D)
__global__ void transpose_w_kernel(const float* __restrict__ w, bf16_t* __restrict__ wt) {
    int i = blockIdx.x * blockDim.x + threadIdx.x;
    if (i >= DN * DN) return;
    int e = i / DN, d = i % DN;
    wt[(size_t)e * DN + d] = (bf16_t)w[(size_t)d * DN + e];
}

// ---------------------------------------------------------------------------
// Stage 1: Y = xb @ W + bias (8192 x 768 x 768). 64x64 tile per wave:
// 4x4 register block, 16 WMMAs per k-step on 16 b128 loads (1 load / WMMA).
// transposed==0: store Y bf16 row-major [row][e]; ==1: store Y^T bf16 [b][e][s]
__global__ void qkv_gemm_kernel(const bf16_t* __restrict__ xb, const bf16_t* __restrict__ wt,
                                const float* __restrict__ bias, bf16_t* __restrict__ out,
                                int transposed) {
    const int lane = threadIdx.x & 31;
    const int wave = threadIdx.x >> 5;
    const int id   = blockIdx.x * 8 + wave;
    const int nN   = DN / 64;                                  // 12
    const int m_blk = id / nN;
    const int n_blk = id % nN;
    if (m_blk >= BN * SN / 64) return;
    const int row0 = m_blk * 64;
    const int e0   = n_blk * 64;

    const bf16_t* Abase = xb + (size_t)row0 * DN;
    const bf16_t* Bbase = wt + (size_t)e0 * DN;

    v8f acc[4][4];
#pragma unroll
    for (int i = 0; i < 4; ++i)
#pragma unroll
        for (int j = 0; j < 4; ++j) acc[i][j] = (v8f){};

    for (int k0 = 0; k0 < DN; k0 += 32) {
        if (k0 + 64 < DN) {
            __builtin_prefetch(Abase + (size_t)(lane & 15) * DN + k0 + 64, 0, 1);
            __builtin_prefetch(Bbase + (size_t)(lane & 15) * DN + k0 + 64, 0, 1);
        }
        v16bf a[4], b[4];
#pragma unroll
        for (int i = 0; i < 4; ++i)
            a[i] = load_frag_a(Abase + (size_t)(16 * i) * DN + k0, DN, lane);
#pragma unroll
        for (int j = 0; j < 4; ++j)
            b[j] = load_frag_b(Bbase + (size_t)(16 * j) * DN + k0, DN, lane);
#pragma unroll
        for (int i = 0; i < 4; ++i)
#pragma unroll
            for (int j = 0; j < 4; ++j)
                acc[i][j] = wmma_bf16(a[i], b[j], acc[i][j]);
    }

    const int half = (lane >> 4) & 1;
    const int n    = lane & 15;
#pragma unroll
    for (int j = 0; j < 4; ++j) {
        const float bn = bias[e0 + 16 * j + n];
        const int   e  = e0 + 16 * j + n;
#pragma unroll
        for (int i = 0; i < 4; ++i) {
#pragma unroll
            for (int r = 0; r < 8; ++r) {
                const int row = row0 + 16 * i + r + half * 8;  // = b*S + s
                const float y = acc[i][j][r] + bn;
                if (!transposed) {
                    out[(size_t)row * DN + e] = (bf16_t)y;
                } else {
                    const int bb = row >> 11;
                    const int ss = row & (SN - 1);
                    out[((size_t)bb * DN + e) * SN + ss] = (bf16_t)y;
                }
            }
        }
    }
}

// ---------------------------------------------------------------------------
// Stage 2: scores = Q K^T / sqrt(D); P = softmax(scores) stored bf16.
// Block: 16 q-rows, 8 waves; wave w covers kv columns [w*256, w*256+128).
// k-outer / t-inner: Q A-fragment loaded once per k-step, reused by 8 WMMAs.
__global__ void attn_softmax_kernel(const bf16_t* __restrict__ Qb, const bf16_t* __restrict__ Kb,
                                    bf16_t* __restrict__ P) {
    __shared__ float red_max[16][8];
    __shared__ float red_sum[16][8];

    const int lane = threadIdx.x & 31;
    const int wave = threadIdx.x >> 5;
    const int rowBase = blockIdx.x * 16;                       // global row = b*S + q
    const int bidx = rowBase >> 11;
    const int half = (lane >> 4) & 1;
    const int n    = lane & 15;
    const float scale = 0.03608439182435161f;                  // 1/sqrt(768)

    v8f c[8];
#pragma unroll
    for (int t = 0; t < 8; ++t) c[t] = (v8f){};

    const bf16_t* Qbase = Qb + (size_t)rowBase * DN;
    const bf16_t* Kbase = Kb + ((size_t)bidx * SN + wave * 256) * DN;

    for (int k0 = 0; k0 < DN; k0 += 32) {
        v16bf a = load_frag_a(Qbase + k0, DN, lane);
#pragma unroll
        for (int t = 0; t < 8; ++t) {
            v16bf b = load_frag_b(Kbase + (size_t)(t * 16) * DN + k0, DN, lane);
            c[t] = wmma_bf16(a, b, c[t]);
        }
    }

    // scale + per-lane row max over the 8 subtiles
    float lm[8];
#pragma unroll
    for (int r = 0; r < 8; ++r) lm[r] = -3.0e38f;
#pragma unroll
    for (int t = 0; t < 8; ++t)
#pragma unroll
        for (int r = 0; r < 8; ++r) {
            c[t][r] *= scale;
            lm[r] = fmaxf(lm[r], c[t][r]);
        }
#pragma unroll
    for (int r = 0; r < 8; ++r) {
        lm[r] = fmaxf(lm[r], __shfl_xor(lm[r], 8, 32));
        lm[r] = fmaxf(lm[r], __shfl_xor(lm[r], 4, 32));
        lm[r] = fmaxf(lm[r], __shfl_xor(lm[r], 2, 32));
        lm[r] = fmaxf(lm[r], __shfl_xor(lm[r], 1, 32));
    }
    if (n == 0) {
#pragma unroll
        for (int r = 0; r < 8; ++r) red_max[r + half * 8][wave] = lm[r];
    }
    __syncthreads();

    float gm[8];
#pragma unroll
    for (int r = 0; r < 8; ++r) {
        const int m = r + half * 8;
        float v = red_max[m][0];
        for (int w = 1; w < 8; ++w) v = fmaxf(v, red_max[m][w]);
        gm[r] = v;
    }

    float ls[8];
#pragma unroll
    for (int r = 0; r < 8; ++r) ls[r] = 0.0f;
#pragma unroll
    for (int t = 0; t < 8; ++t)
#pragma unroll
        for (int r = 0; r < 8; ++r) {
            float e = __expf(c[t][r] - gm[r]);
            c[t][r] = e;
            ls[r] += e;
        }
#pragma unroll
    for (int r = 0; r < 8; ++r) {
        ls[r] += __shfl_xor(ls[r], 8, 32);
        ls[r] += __shfl_xor(ls[r], 4, 32);
        ls[r] += __shfl_xor(ls[r], 2, 32);
        ls[r] += __shfl_xor(ls[r], 1, 32);
    }
    if (n == 0) {
#pragma unroll
        for (int r = 0; r < 8; ++r) red_sum[r + half * 8][wave] = ls[r];
    }
    __syncthreads();

    float inv[8];
#pragma unroll
    for (int r = 0; r < 8; ++r) {
        const int m = r + half * 8;
        float v = 0.0f;
        for (int w = 0; w < 8; ++w) v += red_sum[m][w];
        inv[r] = 1.0f / v;
    }

    for (int t = 0; t < 8; ++t) {
        const int kv0 = wave * 256 + t * 16;
#pragma unroll
        for (int r = 0; r < 8; ++r) {
            const int m = r + half * 8;
            P[(size_t)(rowBase + m) * SN + kv0 + n] = (bf16_t)(c[t][r] * inv[r]);
        }
    }
}

// ---------------------------------------------------------------------------
// Stage 3: O = P @ V. 64x64 tile per wave, 4x4 register block over K=2048.
__global__ void pv_gemm_kernel(const bf16_t* __restrict__ P, const bf16_t* __restrict__ Vt,
                               float* __restrict__ out) {
    const int lane = threadIdx.x & 31;
    const int wave = threadIdx.x >> 5;
    const int id   = blockIdx.x * 8 + wave;
    const int nN   = DN / 64;                                  // 12
    const int m_blk = id / nN;
    const int n_blk = id % nN;
    if (m_blk >= BN * SN / 64) return;
    const int rowBase = m_blk * 64;                            // b*S + q
    const int bidx = rowBase >> 11;
    const int e0   = n_blk * 64;

    const bf16_t* Abase = P  + (size_t)rowBase * SN;
    const bf16_t* Bbase = Vt + ((size_t)bidx * DN + e0) * SN;

    v8f acc[4][4];
#pragma unroll
    for (int i = 0; i < 4; ++i)
#pragma unroll
        for (int j = 0; j < 4; ++j) acc[i][j] = (v8f){};

    for (int k0 = 0; k0 < SN; k0 += 32) {
        if (k0 + 64 < SN) {
            __builtin_prefetch(Abase + (size_t)(lane & 15) * SN + k0 + 64, 0, 1);
            __builtin_prefetch(Bbase + (size_t)(lane & 15) * SN + k0 + 64, 0, 1);
        }
        v16bf a[4], b[4];
#pragma unroll
        for (int i = 0; i < 4; ++i)
            a[i] = load_frag_a(Abase + (size_t)(16 * i) * SN + k0, SN, lane);
#pragma unroll
        for (int j = 0; j < 4; ++j)
            b[j] = load_frag_b(Bbase + (size_t)(16 * j) * SN + k0, SN, lane);
#pragma unroll
        for (int i = 0; i < 4; ++i)
#pragma unroll
            for (int j = 0; j < 4; ++j)
                acc[i][j] = wmma_bf16(a[i], b[j], acc[i][j]);
    }

    const int half = (lane >> 4) & 1;
    const int n    = lane & 15;
#pragma unroll
    for (int i = 0; i < 4; ++i)
#pragma unroll
        for (int j = 0; j < 4; ++j)
#pragma unroll
            for (int r = 0; r < 8; ++r) {
                const int row = rowBase + 16 * i + r + half * 8;
                out[(size_t)row * DN + e0 + 16 * j + n] = acc[i][j][r];
            }
}

// ---------------------------------------------------------------------------
extern "C" void kernel_launch(void* const* d_in, const int* in_sizes, int n_in,
                              void* d_out, int out_size, void* d_ws, size_t ws_size,
                              hipStream_t stream) {
    (void)in_sizes; (void)n_in; (void)out_size; (void)ws_size;
    const float* x  = (const float*)d_in[0];
    const float* Wq = (const float*)d_in[1];
    const float* bq = (const float*)d_in[2];
    const float* Wk = (const float*)d_in[3];
    const float* bk = (const float*)d_in[4];
    const float* Wv = (const float*)d_in[5];
    const float* bv = (const float*)d_in[6];
    float* out = (float*)d_out;

    // workspace carve (bf16 buffers)
    char* ws = (char*)d_ws;
    const size_t xN  = (size_t)BN * SN * DN;       // 6,291,456
    const size_t wN  = (size_t)DN * DN;            // 589,824
    bf16_t* xb  = (bf16_t*)ws;                 ws += xN * 2;
    bf16_t* Wqt = (bf16_t*)ws;                 ws += wN * 2;
    bf16_t* Wkt = (bf16_t*)ws;                 ws += wN * 2;
    bf16_t* Wvt = (bf16_t*)ws;                 ws += wN * 2;
    bf16_t* Qb  = (bf16_t*)ws;                 ws += xN * 2;
    bf16_t* Kb  = (bf16_t*)ws;                 ws += xN * 2;
    bf16_t* Vt  = (bf16_t*)ws;                 ws += xN * 2;
    bf16_t* P   = (bf16_t*)ws;                 // B*S*S*2 = 33.5 MB

    // Stage 0: conversions
    cvt_bf16_kernel<<<2048, 256, 0, stream>>>(x, xb, (int)xN);
    const int wBlocks = (int)((wN + 255) / 256);
    transpose_w_kernel<<<wBlocks, 256, 0, stream>>>(Wq, Wqt);
    transpose_w_kernel<<<wBlocks, 256, 0, stream>>>(Wk, Wkt);
    transpose_w_kernel<<<wBlocks, 256, 0, stream>>>(Wv, Wvt);

    // Stage 1: Q, K, V^T projections. 64x64 wave tiles: 128*12 = 1536 tiles -> 192 blocks.
    const int gemmBlocks = (BN * SN / 64) * (DN / 64) / 8;     // 192
    qkv_gemm_kernel<<<gemmBlocks, 256, 0, stream>>>(xb, Wqt, bq, Qb, 0);
    qkv_gemm_kernel<<<gemmBlocks, 256, 0, stream>>>(xb, Wkt, bk, Kb, 0);
    qkv_gemm_kernel<<<gemmBlocks, 256, 0, stream>>>(xb, Wvt, bv, Vt, 1);

    // Stage 2: softmax(QK^T/sqrt(D)) -> P
    attn_softmax_kernel<<<BN * SN / 16, 256, 0, stream>>>(Qb, Kb, P);

    // Stage 3: O = P V
    pv_gemm_kernel<<<gemmBlocks, 256, 0, stream>>>(P, Vt, out);
}
